// MultiHeadAttention_38233798869424
// MI455X (gfx1250) — compile-verified, matrix operand order checked by reading, not run on hardware
//
#include <hip/hip_runtime.h>

// MHA: B=4, S=2048, IN=OUT=1024, H=16, D=64.
// All matmuls via v_wmma_f32_16x16x32_bf16 (fp32 accumulate).
// Q/K/V/ctx kept as bf16 in d_ws (64 MB, fits in 192MB L2 -> compute bound).
// Q pre-scaled by 1/sqrt(D) in its projection GEMM; V stored head-transposed
// so attention K/V tiles are straight b128 copies -> async loads to LDS.

typedef __bf16 bf16_t;
typedef __attribute__((ext_vector_type(16))) __bf16 v16bf;
typedef __attribute__((ext_vector_type(8)))  __bf16 v8bf;
typedef __attribute__((ext_vector_type(8)))  float  v8f;

#define S_LEN 2048
#define HDIM  64

// ---------------------------------------------------------------------------
// Fragment / WMMA helpers
// ---------------------------------------------------------------------------
__device__ __forceinline__ v16bf ldsfrag(const bf16_t* rowk, int g) {
  // CDNA5 16-bit A/B layout: lane group g holds K in [8g,8g+8) and [16+8g,24+8g).
  union { v16bf v; v8bf h[2]; } u;
  u.h[0] = *reinterpret_cast<const v8bf*>(rowk + 8 * g);
  u.h[1] = *reinterpret_cast<const v8bf*>(rowk + 16 + 8 * g);
  return u.v;
}

__device__ __forceinline__ v8f wmma_bf16(v16bf a, v16bf b, v8f c) {
  return __builtin_amdgcn_wmma_f32_16x16x32_bf16(
      false, a, false, b, (short)0, c, false, false);
}

// ---------------------------------------------------------------------------
// Async global -> LDS copy (16 bytes / lane), ASYNCcnt tracked.
// Builtin signature (probe-confirmed): (v4i AS1*, v4i AS3*, imm off, imm cpol).
// ---------------------------------------------------------------------------
typedef int v4i_g __attribute__((vector_size(16)));
typedef __attribute__((address_space(1))) v4i_g* gas_v4i;
typedef __attribute__((address_space(3))) v4i_g* las_v4i;

__device__ __forceinline__ void async_copy16(bf16_t* lds_dst, const bf16_t* g_src) {
#if __has_builtin(__builtin_amdgcn_global_load_async_to_lds_b128)
  __builtin_amdgcn_global_load_async_to_lds_b128(
      (gas_v4i)(uintptr_t)g_src,
      (las_v4i)(unsigned)(uintptr_t)lds_dst,
      0, 0);
#else
  *reinterpret_cast<v8bf*>(lds_dst) = *reinterpret_cast<const v8bf*>(g_src);
#endif
}

__device__ __forceinline__ void wait_async0() {
#if __has_builtin(__builtin_amdgcn_global_load_async_to_lds_b128)
#if __has_builtin(__builtin_amdgcn_s_wait_asynccnt)
  __builtin_amdgcn_s_wait_asynccnt(0);
#else
  asm volatile("s_wait_asynccnt 0x0" ::: "memory");
#endif
#endif
}

// ---------------------------------------------------------------------------
// Reductions across each 16-lane group (wave32): v_permlane16 (VALU) if
// available, else shfl (ds_bpermute) fallback.
// ---------------------------------------------------------------------------
#if __has_builtin(__builtin_amdgcn_permlane16)
__device__ __forceinline__ float xor16_lane(float v, int lo, int hi) {
  int x = __float_as_int(v);
  return __int_as_float(__builtin_amdgcn_permlane16(x, x, lo, hi, false, false));
}
__device__ __forceinline__ float red16_max(float v) {
  v = fmaxf(v, xor16_lane(v, 0x67452301, (int)0xEFCDAB89));  // xor 1
  v = fmaxf(v, xor16_lane(v, 0x54761032, (int)0xDCFE98BA));  // xor 2
  v = fmaxf(v, xor16_lane(v, 0x32107654, (int)0xBA98FEDC));  // xor 4
  v = fmaxf(v, xor16_lane(v, (int)0xFEDCBA98, 0x76543210));  // xor 8
  return v;
}
__device__ __forceinline__ float red16_sum(float v) {
  v += xor16_lane(v, 0x67452301, (int)0xEFCDAB89);
  v += xor16_lane(v, 0x54761032, (int)0xDCFE98BA);
  v += xor16_lane(v, 0x32107654, (int)0xBA98FEDC);
  v += xor16_lane(v, (int)0xFEDCBA98, 0x76543210);
  return v;
}
#else
__device__ __forceinline__ float red16_max(float v) {
  v = fmaxf(v, __shfl_xor(v, 1)); v = fmaxf(v, __shfl_xor(v, 2));
  v = fmaxf(v, __shfl_xor(v, 4)); v = fmaxf(v, __shfl_xor(v, 8));
  return v;
}
__device__ __forceinline__ float red16_sum(float v) {
  v += __shfl_xor(v, 1); v += __shfl_xor(v, 2);
  v += __shfl_xor(v, 4); v += __shfl_xor(v, 8);
  return v;
}
#endif

// ---------------------------------------------------------------------------
// Tiled GEMM: C[M,N] = oscale * (A[M,K] * Bm[K,N]) (+ bias).
// Block 128x128, 8 waves (4 M x 2 N), wave tile 32x64, K step 32.
// VT=true stores output in per-head transposed layout [BH][D][S] (for V).
// ---------------------------------------------------------------------------
template <typename AT, typename CT, bool BIAS, bool VT>
__global__ __launch_bounds__(256) void gemm_bf16_wmma(
    const AT* __restrict__ A, const float* __restrict__ Bm,
    const float* __restrict__ bias, CT* __restrict__ C,
    int M, int N, int K, float oscale) {
  __shared__ alignas(16) bf16_t As[128 * 40];  // [m][k]
  __shared__ alignas(16) bf16_t Bs[128 * 40];  // transposed: [n][k]

  const int tid  = threadIdx.x;
  const int lane = tid & 31, wave = tid >> 5;
  const int ln = lane & 15, g = lane >> 4;
  const int wm = wave & 3, wn = wave >> 2;
  const long row0 = (long)blockIdx.y * 128;
  const long col0 = (long)blockIdx.x * 128;

  v8f acc[2][4];
#pragma unroll
  for (int mi = 0; mi < 2; ++mi)
#pragma unroll
    for (int ni = 0; ni < 4; ++ni)
      acc[mi][ni] = v8f{0.f, 0.f, 0.f, 0.f, 0.f, 0.f, 0.f, 0.f};

  for (int k0 = 0; k0 < K; k0 += 32) {
    __syncthreads();
#pragma unroll
    for (int i = 0; i < 16; ++i) {       // A tile 128x32 (convert to bf16)
      int idx = tid + i * 256;
      int r = idx >> 5, c = idx & 31;
      As[r * 40 + c] = (bf16_t)(float)A[(row0 + r) * K + (k0 + c)];
    }
#pragma unroll
    for (int i = 0; i < 16; ++i) {       // B tile 32x128 -> [n][k]
      int idx = tid + i * 256;
      int n = idx & 127, c = idx >> 7;
      Bs[n * 40 + c] = (bf16_t)Bm[(long)(k0 + c) * N + (col0 + n)];
    }
    __syncthreads();

    v16bf af[2], bfm[4];
#pragma unroll
    for (int mi = 0; mi < 2; ++mi)
      af[mi] = ldsfrag(As + (wm * 32 + mi * 16 + ln) * 40, g);
#pragma unroll
    for (int ni = 0; ni < 4; ++ni)
      bfm[ni] = ldsfrag(Bs + (wn * 64 + ni * 16 + ln) * 40, g);
#pragma unroll
    for (int mi = 0; mi < 2; ++mi)
#pragma unroll
      for (int ni = 0; ni < 4; ++ni)
        acc[mi][ni] = wmma_bf16(af[mi], bfm[ni], acc[mi][ni]);
  }

  // C/D layout: lane holds col = ln, rows r + 8g.
#pragma unroll
  for (int mi = 0; mi < 2; ++mi)
#pragma unroll
    for (int ni = 0; ni < 4; ++ni)
#pragma unroll
      for (int r = 0; r < 8; ++r) {
        long row = row0 + wm * 32 + mi * 16 + r + 8 * g;
        long col = col0 + wn * 64 + ni * 16 + ln;
        float v = acc[mi][ni][r] * oscale;
        if (BIAS) v += bias[col];
        if (VT) {
          // faithful view: b=row/2048, h=(row%2048)/128,
          // s2=(row%128)*16 + col/64, d=col%64 -> store [bh][d][s2]
          long b  = row >> 11;
          long il = row & 2047;
          long h  = il >> 7;
          long s2 = ((il & 127) << 4) + (col >> 6);
          long d  = col & 63;
          C[(((b << 4) + h) * HDIM + d) * (long)S_LEN + s2] = (CT)v;
        } else {
          C[row * N + col] = (CT)v;
        }
      }
}

// ---------------------------------------------------------------------------
// Flash attention per (b,h). Q,K: contiguous [2048,64] bf16 (Q pre-scaled);
// Vt: head-transposed [64,2048] bf16. Block = 128 query rows (16/wave),
// key blocks of 64, online softmax, K/V tiles via async loads to LDS.
// ---------------------------------------------------------------------------
__global__ __launch_bounds__(256) void flash_attn(
    const bf16_t* __restrict__ Q, const bf16_t* __restrict__ K,
    const bf16_t* __restrict__ Vt, bf16_t* __restrict__ Ctx) {
  __shared__ alignas(16) bf16_t Ks[64 * 72];     // [key][d]  (B^T layout for Q*K^T)
  __shared__ alignas(16) bf16_t Vs[64 * 72];     // [d][key]  (B^T layout for P*V)
  __shared__ alignas(16) bf16_t Ps[8][16 * 72];  // per-wave P tile, [m][key]

  const int tid = threadIdx.x, wave = tid >> 5, lane = tid & 31;
  const int ln = lane & 15, g = lane >> 4;
  const long head_off = (long)blockIdx.y * S_LEN * HDIM;
  const bf16_t* Qh  = Q  + head_off;
  const bf16_t* Kh  = K  + head_off;
  const bf16_t* Vth = Vt + head_off;   // [64][2048]
  bf16_t* Ch = Ctx + head_off;
  const int q0 = blockIdx.x * 128 + wave * 16;

  // Q fragments (A layout) straight from global: row m = q0+ln.
  v16bf qa[2];
#pragma unroll
  for (int kc = 0; kc < 2; ++kc) {
    union { v16bf v; v8bf h[2]; } u;
    const bf16_t* base = Qh + (long)(q0 + ln) * HDIM + kc * 32;
    u.h[0] = *reinterpret_cast<const v8bf*>(base + 8 * g);
    u.h[1] = *reinterpret_cast<const v8bf*>(base + 16 + 8 * g);
    qa[kc] = u.v;
  }

  v8f o[4];
  float m_i[8], l_i[8];
#pragma unroll
  for (int n = 0; n < 4; ++n) o[n] = v8f{0.f, 0.f, 0.f, 0.f, 0.f, 0.f, 0.f, 0.f};
#pragma unroll
  for (int r = 0; r < 8; ++r) { m_i[r] = -__builtin_inff(); l_i[r] = 0.f; }

  for (int kb = 0; kb < S_LEN; kb += 64) {
    __syncthreads();
    // Async copy 64x64 K tile and 64x64 V^T tile (16B / lane / chunk).
#pragma unroll
    for (int i = 0; i < 2; ++i) {
      int e = (tid * 2 + i) * 8;   // covers 4096 elements
      int r = e >> 6, c = e & 63;
      async_copy16(Ks + r * 72 + c, Kh  + (long)(kb + r) * HDIM + c);
      async_copy16(Vs + r * 72 + c, Vth + (long)r * S_LEN + kb + c);
    }
    wait_async0();
    __syncthreads();

    // S = Q*K^T (Q pre-scaled): 16x64 per wave = 4 C-frags.
    v8f s[4];
#pragma unroll
    for (int n = 0; n < 4; ++n) {
      s[n] = v8f{0.f, 0.f, 0.f, 0.f, 0.f, 0.f, 0.f, 0.f};
#pragma unroll
      for (int kc = 0; kc < 2; ++kc)
        s[n] = wmma_bf16(qa[kc], ldsfrag(Ks + (n * 16 + ln) * 72 + kc * 32, g), s[n]);
    }

    // Online softmax: each lane owns rows r+8g; reduce over 16 cols/group.
#pragma unroll
    for (int r = 0; r < 8; ++r) {
      float pm = fmaxf(fmaxf(s[0][r], s[1][r]), fmaxf(s[2][r], s[3][r]));
      pm = red16_max(pm);
      float m_new = fmaxf(m_i[r], pm);
      float alpha = __expf(m_i[r] - m_new);
      m_i[r] = m_new;
      float rs = 0.f;
#pragma unroll
      for (int n = 0; n < 4; ++n) {
        float p = __expf(s[n][r] - m_new);
        s[n][r] = p;
        rs += p;
      }
      rs = red16_sum(rs);
      l_i[r] = l_i[r] * alpha + rs;
#pragma unroll
      for (int n = 0; n < 4; ++n) o[n][r] *= alpha;
    }

    // C-layout -> A-layout for P via per-wave LDS scratch.
    bf16_t* pw = &Ps[wave][0];
#pragma unroll
    for (int n = 0; n < 4; ++n)
#pragma unroll
      for (int r = 0; r < 8; ++r)
        pw[(r + 8 * g) * 72 + n * 16 + ln] = (bf16_t)s[n][r];

    v16bf pa[2];
#pragma unroll
    for (int kc = 0; kc < 2; ++kc)
      pa[kc] = ldsfrag(pw + ln * 72 + kc * 32, g);

    // O += P * V.
#pragma unroll
    for (int n = 0; n < 4; ++n)
#pragma unroll
      for (int kc = 0; kc < 2; ++kc)
        o[n] = wmma_bf16(pa[kc], ldsfrag(Vs + (n * 16 + ln) * 72 + kc * 32, g), o[n]);
  }

  // Normalize and store ctx (bf16) in the contiguous [S,D] head slice.
#pragma unroll
  for (int r = 0; r < 8; ++r) {
    float inv = 1.f / l_i[r];
#pragma unroll
    for (int n = 0; n < 4; ++n) {
      long row = q0 + r + 8 * g;
      Ch[row * HDIM + n * 16 + ln] = (bf16_t)(o[n][r] * inv);
    }
  }
}

// ---------------------------------------------------------------------------
extern "C" void kernel_launch(void* const* d_in, const int* in_sizes, int n_in,
                              void* d_out, int out_size, void* d_ws, size_t ws_size,
                              hipStream_t stream) {
  const float* x  = (const float*)d_in[0];
  const float* Wq = (const float*)d_in[1];
  const float* Wk = (const float*)d_in[2];
  const float* Wv = (const float*)d_in[3];
  const float* Wp = (const float*)d_in[4];
  const float* bp = (const float*)d_in[5];
  float* out = (float*)d_out;

  const int M = 4 * S_LEN;          // 8192
  const int N = 1024, Kd = 1024;
  const long elems = (long)M * N;   // 8M per buffer

  bf16_t* Qb = (bf16_t*)d_ws;
  bf16_t* Kb = Qb + elems;
  bf16_t* Vb = Kb + elems;          // head-transposed [BH][64][2048]
  bf16_t* Cb = Vb + elems;          // 64 MB total bf16 workspace

  dim3 block(256);
  dim3 gridG(N / 128, M / 128);     // (8, 64)
  dim3 gridA(S_LEN / 128, 4 * 16);  // (16, 64)

  const float qscale = 0.125f;      // 1/sqrt(64) folded into Q projection

  gemm_bf16_wmma<float, bf16_t, false, false><<<gridG, block, 0, stream>>>(
      x, Wq, nullptr, Qb, M, N, Kd, qscale);
  gemm_bf16_wmma<float, bf16_t, false, false><<<gridG, block, 0, stream>>>(
      x, Wk, nullptr, Kb, M, N, Kd, 1.0f);
  gemm_bf16_wmma<float, bf16_t, false, true><<<gridG, block, 0, stream>>>(
      x, Wv, nullptr, Vb, M, N, Kd, 1.0f);
  flash_attn<<<gridA, block, 0, stream>>>(Qb, Kb, Vb, Cb);
  gemm_bf16_wmma<bf16_t, float, true, false><<<gridG, block, 0, stream>>>(
      Cb, Wp, bp, out, M, N, Kd, 1.0f);
}
